// MultiHeadAttention_18073222382184
// MI455X (gfx1250) — compile-verified, matrix operand order checked by reading, not run on hardware
//
#include <hip/hip_runtime.h>

#define EMBED   768
#define NHEADS  12
#define HDIM    64
#define SEQ     4096
#define BATCH   2
#define QKVN    (3 * EMBED)      // 2304
#define MTOT    (BATCH * SEQ)    // 8192
#define ATTN_SCALE 0.125f        // 64^-0.5
#define NKSTEPS (EMBED / 32)     // 24

typedef __attribute__((ext_vector_type(16))) _Float16 v16h;
typedef __attribute__((ext_vector_type(8)))  _Float16 v8h;
typedef __attribute__((ext_vector_type(8)))  float    v8f;
typedef __attribute__((ext_vector_type(4)))  unsigned u32x4;
typedef __attribute__((ext_vector_type(8)))  unsigned u32x8;

// Build a 16-half WMMA operand fragment per the CDNA5 wave32 16-bit layout:
// lanes 0-15 hold K = {kbase..+7, kbase+16..+23} with kbase=0, lanes 16-31
// the same with kbase=8.  Two 16-byte loads per lane.
static __device__ __forceinline__ v16h frag_ld(const _Float16* p, int kbase) {
    v8h lo = *(const v8h*)(p + kbase);
    v8h hi = *(const v8h*)(p + kbase + 16);
    v16h r;
#pragma unroll
    for (int i = 0; i < 8; ++i) { r[i] = lo[i]; r[i + 8] = hi[i]; }
    return r;
}

static __device__ __forceinline__ v8f wmma_f16(v16h a, v16h b, v8f c) {
    return __builtin_amdgcn_wmma_f32_16x16x32_f16(
        false, a, false, b, (short)0, c, false, false);
}

// Async global -> LDS 16-byte copy (CDNA5, tracked by ASYNCcnt).
static __device__ __forceinline__ void async_b128(unsigned lds_off, const _Float16* gsrc) {
    asm volatile("global_load_async_to_lds_b128 %0, %1, off"
                 :: "v"(lds_off), "v"((unsigned long long)(uintptr_t)gsrc)
                 : "memory");
}
static __device__ __forceinline__ void wait_async_le1() {
    asm volatile("s_wait_asynccnt 0x1" ::: "memory");
}
static __device__ __forceinline__ void wait_async_le0() {
    asm volatile("s_wait_asynccnt 0x0" ::: "memory");
}

// Tensor Data Mover: one 2D tile load (rows x 32 halfs) global -> LDS.
// D# group0/group1 built in SGPRs per CDNA5 ISA.
// data_size=2B; tile = 32(halfs) x 128(rows); tensor_dim0_stride = 768 halfs;
// LDS pad: 4 DWORDs every 16 DWORDs -> 80-byte padded row stride.
static __device__ __forceinline__ void tdm_load_b_tile(
    unsigned lds_off, const _Float16* gsrc, unsigned rows_total)
{
    const unsigned long long ga = (unsigned long long)(uintptr_t)gsrc;
    u32x4 g0;
    g0[0] = 1u;                                   // count=1 (valid descriptor)
    g0[1] = lds_off;                              // lds_addr [63:32]
    g0[2] = (unsigned)ga;                         // global_addr [95:64]
    g0[3] = (unsigned)(ga >> 32) | (2u << 30);    // global_addr[56:32] | type=2
    u32x8 g1;
    g1[0] = (1u << 16)            // data_size = 1 -> 2 bytes
          | (1u << 20)            // pad_enable
          | (3u << 22)            // pad_interval: 16 DWORDs
          | (3u << 25);           // pad_amount: 4 DWORDs
    g1[1] = (768u & 0xFFFFu) << 16;               // tensor_dim0[15:0] = 768
    g1[2] = (rows_total & 0xFFFFu) << 16;         // tensor_dim1[15:0]
    g1[3] = (rows_total >> 16) | (32u << 16);     // tensor_dim1[31:16] | tile_dim0=32
    g1[4] = 128u;                                 // tile_dim1=128, tile_dim2=0
    g1[5] = 768u;                                 // tensor_dim0_stride = 768
    g1[6] = 0u;
    g1[7] = 0u;
    asm volatile("tensor_load_to_lds %0, %1"
                 :: "s"(g0), "s"(g1)
                 : "memory");
}

// ---------------------------------------------------------------------------
// One-shot precision/layout prep kernels (tiny vs. the GEMM cost).
// ---------------------------------------------------------------------------
__global__ __launch_bounds__(256) void cvt_f32_to_f16_kernel(
    const float* __restrict__ src, _Float16* __restrict__ dst, int n)
{
    const int i = (blockIdx.x * 256 + threadIdx.x) * 8;
    if (i + 8 <= n) {
        float4 a = *(const float4*)(src + i);
        float4 b = *(const float4*)(src + i + 4);
        v8h o;
        o[0] = (_Float16)a.x; o[1] = (_Float16)a.y; o[2] = (_Float16)a.z; o[3] = (_Float16)a.w;
        o[4] = (_Float16)b.x; o[5] = (_Float16)b.y; o[6] = (_Float16)b.z; o[7] = (_Float16)b.w;
        *(v8h*)(dst + i) = o;
    }
}

// dst[n][k] = (f16)src[k][n]  (weights -> transposed f16 so every WMMA B
// fragment is a contiguous 16B load).  32x32 LDS tile transpose.
__global__ __launch_bounds__(256) void transpose_to_f16_kernel(
    const float* __restrict__ src, _Float16* __restrict__ dst, int K, int N)
{
    __shared__ float tile[32][33];
    const int k0 = blockIdx.y * 32, n0 = blockIdx.x * 32;
    const int tx = threadIdx.x & 31, ty = threadIdx.x >> 5;
#pragma unroll
    for (int i = 0; i < 4; ++i) {
        const int k = ty + i * 8;
        tile[k][tx] = src[(size_t)(k0 + k) * N + n0 + tx];
    }
    __syncthreads();
#pragma unroll
    for (int i = 0; i < 4; ++i) {
        const int n = ty + i * 8;
        dst[(size_t)(n0 + n) * K + k0 + tx] = (_Float16)tile[tx][n];
    }
}

// Scatter one QKV-projection result element into head-major Q/K and transposed V.
static __device__ __forceinline__ void scatter_qkv(
    float val, int m, int n,
    _Float16* __restrict__ qb, _Float16* __restrict__ kb, _Float16* __restrict__ vtb)
{
    const int bi = m >> 12;             // m / SEQ
    const int s  = m & (SEQ - 1);
    const int which = n / EMBED;        // 0=Q 1=K 2=V
    const int rem = n - which * EMBED;
    const int h = rem >> 6;
    const int d = rem & (HDIM - 1);
    const _Float16 hv = (_Float16)val;
    const size_t bh = (size_t)bi * NHEADS + h;
    if (which == 0)      qb [(bh * SEQ + s) * HDIM + d] = hv;   // [B,H,S,64]
    else if (which == 1) kb [(bh * SEQ + s) * HDIM + d] = hv;   // [B,H,S,64]
    else                 vtb[(bh * HDIM + d) * SEQ + s] = hv;   // [B,H,64,S]
}

// ---------------------------------------------------------------------------
// Kernel: QKV projection.  C[8192 x 2304] = xh[8192 x 768] * wqT^T + b.
// Block tile 64(M) x 128(N), 8 waves, each wave 16x64 (4 WMMA acc).
// Double-buffered LDS pipeline: A tile via async-to-LDS (1 op/thread/step),
// B tile via one TDM tensor_load_to_lds per step (wave 0 issues).
// ---------------------------------------------------------------------------
__global__ __launch_bounds__(256) void qkv_proj_kernel(
    const _Float16* __restrict__ xh, const _Float16* __restrict__ wqT,
    const float* __restrict__ bias,
    _Float16* __restrict__ qb, _Float16* __restrict__ kbuf, _Float16* __restrict__ vtb)
{
    __shared__ __align__(16) _Float16 lx[2][64][40];    // A tiles [m][k]
    __shared__ __align__(16) _Float16 lw[2][128][40];   // B tiles [n][k]
    const int t = threadIdx.x;
    const int lane = t & 31, wave = t >> 5;
    const int mw = wave & 3;          // 16-row subtile
    const int nw = wave >> 2;         // 64-col half
    const int n0 = blockIdx.x * 128;
    const int m0 = blockIdx.y * 64;
    const int kbase = (lane & 16) ? 8 : 0;
    const int nl = lane & 15;
    const unsigned lxb = (unsigned)(uintptr_t)&lx[0][0][0];
    const unsigned lwb = (unsigned)(uintptr_t)&lw[0][0][0];
    const int ar = t >> 2, ack = (t & 3) * 8;   // per-thread A-stage chunk

    v8f acc[4] = {};

    // prologue: stage tile 0 into buffer 0
    async_b128(lxb + (unsigned)(ar * 80 + ack * 2),
               xh + (size_t)(m0 + ar) * EMBED + ack);
    if (wave == 0)
        tdm_load_b_tile(lwb, wqT + (size_t)n0 * EMBED, QKVN);

    for (int k = 0; k < NKSTEPS; ++k) {
        const int cur = k & 1;
        if (k + 1 < NKSTEPS) {
            const int k0n = (k + 1) * 32;
            async_b128(lxb + (unsigned)((cur ^ 1) * 5120 + ar * 80 + ack * 2),
                       xh + (size_t)(m0 + ar) * EMBED + k0n + ack);
            if (wave == 0)
                tdm_load_b_tile(lwb + (unsigned)((cur ^ 1) * 10240),
                                wqT + (size_t)n0 * EMBED + k0n, QKVN);
            wait_async_le1();
            if (wave == 0) __builtin_amdgcn_s_wait_tensorcnt(1);
        } else {
            wait_async_le0();
            if (wave == 0) __builtin_amdgcn_s_wait_tensorcnt(0);
        }
        __syncthreads();                 // tile k visible to all waves
        const v16h a = frag_ld(&lx[cur][mw * 16 + nl][0], kbase);
#pragma unroll
        for (int j = 0; j < 4; ++j)
            acc[j] = wmma_f16(a, frag_ld(&lw[cur][nw * 64 + j * 16 + nl][0], kbase), acc[j]);
        __syncthreads();                 // all waves done reading buffer `cur`
    }

    const int rbase = m0 + mw * 16 + ((lane & 16) ? 8 : 0);
#pragma unroll
    for (int j = 0; j < 4; ++j) {
        const int ncol = n0 + nw * 64 + j * 16 + nl;
        const float bv = bias[ncol];
#pragma unroll
        for (int r = 0; r < 8; ++r)
            scatter_qkv(acc[j][r] + bv, rbase + r, ncol, qb, kbuf, vtb);
    }
}

// ---------------------------------------------------------------------------
// Kernel: flash attention.  One wave = 16 Q rows; KV streamed in tiles of 32
// with online softmax; P round-trips per-wave LDS to become an A-fragment.
// ---------------------------------------------------------------------------
__global__ __launch_bounds__(128) void flash_attn_kernel(
    const _Float16* __restrict__ qb, const _Float16* __restrict__ kb,
    const _Float16* __restrict__ vtb, _Float16* __restrict__ attn)
{
    __shared__ __align__(16) _Float16 ldsP[4][16 * 32];   // 1KB per wave
    const int t = threadIdx.x, lane = t & 31, wave = t >> 5;
    const int b = blockIdx.z, h = blockIdx.y;
    const int q0 = blockIdx.x * 64 + wave * 16;
    const int kbase = (lane & 16) ? 8 : 0;
    const int nl = lane & 15;
    const size_t bh = (size_t)b * NHEADS + h;
    const _Float16* qh = qb  + bh * SEQ * HDIM;
    const _Float16* kh = kb  + bh * SEQ * HDIM;
    const _Float16* vh = vtb + bh * HDIM * SEQ;   // [d][s]
    _Float16* Pw = &ldsP[wave][0];

    const _Float16* qp = qh + (size_t)(q0 + nl) * HDIM;
    const v16h qa0 = frag_ld(qp, kbase);
    const v16h qa1 = frag_ld(qp + 32, kbase);

    v8f o0 = {}, o1 = {}, o2 = {}, o3 = {};
    float mrow[8], lrow[8];
#pragma unroll
    for (int r = 0; r < 8; ++r) { mrow[r] = -3.0e38f; lrow[r] = 0.0f; }

    for (int kv0 = 0; kv0 < SEQ; kv0 += 32) {
        if (kv0 + 32 < SEQ) {   // prefetch next KV tile (global_prefetch_b8)
            __builtin_prefetch(kh + (size_t)(kv0 + 32 + nl) * HDIM, 0, 0);
            __builtin_prefetch(vh + (size_t)nl * SEQ + kv0 + 32, 0, 0);
        }
        const _Float16* kp0 = kh + (size_t)(kv0 + nl) * HDIM;
        const _Float16* kp1 = kh + (size_t)(kv0 + 16 + nl) * HDIM;
        v8f s0 = {}, s1 = {};
        s0 = wmma_f16(qa0, frag_ld(kp0,      kbase), s0);
        s0 = wmma_f16(qa1, frag_ld(kp0 + 32, kbase), s0);
        s1 = wmma_f16(qa0, frag_ld(kp1,      kbase), s1);
        s1 = wmma_f16(qa1, frag_ld(kp1 + 32, kbase), s1);

        float alpha[8];
#pragma unroll
        for (int r = 0; r < 8; ++r) {
            const float x0 = s0[r] * ATTN_SCALE;
            const float x1 = s1[r] * ATTN_SCALE;
            float cand = fmaxf(x0, x1);
#pragma unroll
            for (int msk = 1; msk < 16; msk <<= 1)
                cand = fmaxf(cand, __shfl_xor(cand, msk, 32));
            const float mn = fmaxf(mrow[r], cand);
            alpha[r] = __expf(mrow[r] - mn);
            mrow[r] = mn;
            const float p0 = __expf(x0 - mn);
            const float p1 = __expf(x1 - mn);
            float rs = p0 + p1;
#pragma unroll
            for (int msk = 1; msk < 16; msk <<= 1)
                rs += __shfl_xor(rs, msk, 32);
            lrow[r] = lrow[r] * alpha[r] + rs;
            const int row = r + ((lane & 16) ? 8 : 0);
            Pw[row * 32 + nl]      = (_Float16)p0;
            Pw[row * 32 + 16 + nl] = (_Float16)p1;
        }
#pragma unroll
        for (int r = 0; r < 8; ++r) {
            o0[r] *= alpha[r]; o1[r] *= alpha[r];
            o2[r] *= alpha[r]; o3[r] *= alpha[r];
        }
        asm volatile("s_wait_dscnt 0" ::: "memory");   // intra-wave LDS RAW fence
        const v16h pa = frag_ld(Pw + nl * 32, kbase);
        const _Float16* vp = vh + kv0;
        o0 = wmma_f16(pa, frag_ld(vp + (size_t)(nl)      * SEQ, kbase), o0);
        o1 = wmma_f16(pa, frag_ld(vp + (size_t)(16 + nl) * SEQ, kbase), o1);
        o2 = wmma_f16(pa, frag_ld(vp + (size_t)(32 + nl) * SEQ, kbase), o2);
        o3 = wmma_f16(pa, frag_ld(vp + (size_t)(48 + nl) * SEQ, kbase), o3);
    }

#pragma unroll
    for (int r = 0; r < 8; ++r) {
        const float inv = 1.0f / lrow[r];
        const int row = q0 + r + ((lane & 16) ? 8 : 0);
        const size_t base = ((size_t)b * SEQ + row) * EMBED + (size_t)h * HDIM;
        attn[base + 0 * 16 + nl] = (_Float16)(o0[r] * inv);
        attn[base + 1 * 16 + nl] = (_Float16)(o1[r] * inv);
        attn[base + 2 * 16 + nl] = (_Float16)(o2[r] * inv);
        attn[base + 3 * 16 + nl] = (_Float16)(o3[r] * inv);
    }
}

// ---------------------------------------------------------------------------
// Kernel: output projection.  out[8192 x 768] = attn_f16 @ woT^T + b (fp32).
// Same double-buffered async+TDM pipeline as the QKV GEMM.
// ---------------------------------------------------------------------------
__global__ __launch_bounds__(256) void out_proj_kernel(
    const _Float16* __restrict__ attn, const _Float16* __restrict__ woT,
    const float* __restrict__ bias, float* __restrict__ out)
{
    __shared__ __align__(16) _Float16 lx[2][64][40];
    __shared__ __align__(16) _Float16 lw[2][128][40];
    const int t = threadIdx.x;
    const int lane = t & 31, wave = t >> 5;
    const int mw = wave & 3, nw = wave >> 2;
    const int n0 = blockIdx.x * 128;
    const int m0 = blockIdx.y * 64;
    const int kbase = (lane & 16) ? 8 : 0;
    const int nl = lane & 15;
    const unsigned lxb = (unsigned)(uintptr_t)&lx[0][0][0];
    const unsigned lwb = (unsigned)(uintptr_t)&lw[0][0][0];
    const int ar = t >> 2, ack = (t & 3) * 8;

    v8f acc[4] = {};

    async_b128(lxb + (unsigned)(ar * 80 + ack * 2),
               attn + (size_t)(m0 + ar) * EMBED + ack);
    if (wave == 0)
        tdm_load_b_tile(lwb, woT + (size_t)n0 * EMBED, EMBED);

    for (int k = 0; k < NKSTEPS; ++k) {
        const int cur = k & 1;
        if (k + 1 < NKSTEPS) {
            const int k0n = (k + 1) * 32;
            async_b128(lxb + (unsigned)((cur ^ 1) * 5120 + ar * 80 + ack * 2),
                       attn + (size_t)(m0 + ar) * EMBED + k0n + ack);
            if (wave == 0)
                tdm_load_b_tile(lwb + (unsigned)((cur ^ 1) * 10240),
                                woT + (size_t)n0 * EMBED + k0n, EMBED);
            wait_async_le1();
            if (wave == 0) __builtin_amdgcn_s_wait_tensorcnt(1);
        } else {
            wait_async_le0();
            if (wave == 0) __builtin_amdgcn_s_wait_tensorcnt(0);
        }
        __syncthreads();
        const v16h a = frag_ld(&lx[cur][mw * 16 + nl][0], kbase);
#pragma unroll
        for (int j = 0; j < 4; ++j)
            acc[j] = wmma_f16(a, frag_ld(&lw[cur][nw * 64 + j * 16 + nl][0], kbase), acc[j]);
        __syncthreads();
    }

    const int rbase = m0 + mw * 16 + ((lane & 16) ? 8 : 0);
#pragma unroll
    for (int j = 0; j < 4; ++j) {
        const int ncol = n0 + nw * 64 + j * 16 + nl;
        const float bv = bias[ncol];
#pragma unroll
        for (int r = 0; r < 8; ++r)
            out[(size_t)(rbase + r) * EMBED + ncol] = acc[j][r] + bv;
    }
}

extern "C" void kernel_launch(void* const* d_in, const int* in_sizes, int n_in,
                              void* d_out, int out_size, void* d_ws, size_t ws_size,
                              hipStream_t stream) {
    const float* x     = (const float*)d_in[0];
    const float* w_qkv = (const float*)d_in[1];
    const float* b_qkv = (const float*)d_in[2];
    const float* w_out = (const float*)d_in[3];
    const float* b_out = (const float*)d_in[4];
    float* out = (float*)d_out;

    // Workspace layout (all f16):
    // xh [8192,768] | wqT [2304,768] | woT [768,768] |
    // Q [B,H,S,64] | K [B,H,S,64] | V^T [B,H,64,S] | attn [8192,768]
    const size_t XH  = (size_t)MTOT * EMBED;
    const size_t WQT = (size_t)QKVN * EMBED;
    const size_t WOT = (size_t)EMBED * EMBED;
    const size_t HE  = (size_t)BATCH * NHEADS * SEQ * HDIM;
    _Float16* xh   = (_Float16*)d_ws;
    _Float16* wqT  = xh  + XH;
    _Float16* woT  = wqT + WQT;
    _Float16* qbuf = woT + WOT;
    _Float16* kbuf = qbuf + HE;
    _Float16* vtb  = kbuf + HE;
    _Float16* attn = vtb  + HE;     // ~68 MB total

    cvt_f32_to_f16_kernel<<<(int)(XH / 2048), 256, 0, stream>>>(x, xh, (int)XH);
    transpose_to_f16_kernel<<<dim3(QKVN / 32, EMBED / 32), 256, 0, stream>>>(
        w_qkv, wqT, EMBED, QKVN);
    transpose_to_f16_kernel<<<dim3(EMBED / 32, EMBED / 32), 256, 0, stream>>>(
        w_out, woT, EMBED, EMBED);
    qkv_proj_kernel<<<dim3(QKVN / 128, MTOT / 64), 256, 0, stream>>>(
        xh, wqT, b_qkv, qbuf, kbuf, vtb);
    flash_attn_kernel<<<dim3(SEQ / 64, NHEADS, BATCH), 128, 0, stream>>>(
        qbuf, kbuf, vtb, attn);
    out_proj_kernel<<<dim3(EMBED / 128, MTOT / 64), 256, 0, stream>>>(
        attn, woT, b_out, out);
}
